// GNNPooling_pyg_11819749998823
// MI455X (gfx1250) — matrix operand order repaired and checked by reading, not compiled, not run on hardware
//
#include <hip/hip_runtime.h>
#include <cmath>

typedef __bf16 bf16_t;
typedef __attribute__((ext_vector_type(16))) __bf16 v16bf;
typedef __attribute__((ext_vector_type(8)))  __bf16 v8bf;
typedef __attribute__((ext_vector_type(8)))  float  v8f;
typedef __attribute__((ext_vector_type(4)))  float  v4f;

#define N_NODES 64
#define DOUT    128

// ---------------------------------------------------------------------------
// Combine W[K][Fin][128] with Chebyshev coefficients:
//   WpT[o][f] = sum_k p_k W[k][f][o]   (bf16, K-major for WMMA B fragments)
//   Wq [f][o] = sum_k q_k W[k][f][o]   (fp32, for the tiny node-sum GEMM)
// ---------------------------------------------------------------------------
__global__ __launch_bounds__(256)
void prep_weights(const float* __restrict__ W, bf16_t* __restrict__ WpT,
                  float* __restrict__ Wq, int Kc, int Fin,
                  float p0, float p1, float p2, float p3, float p4,
                  float q0, float q1, float q2, float q3, float q4)
{
    long t = (long)blockIdx.x * 256 + threadIdx.x;
    if (t >= (long)Fin * DOUT) return;
    int f = (int)(t >> 7);
    int o = (int)(t & 127);
    float pc[5] = {p0, p1, p2, p3, p4};
    float qc[5] = {q0, q1, q2, q3, q4};
    float sp = 0.f, sq = 0.f;
    for (int k = 0; k < Kc; ++k) {
        float w = W[((long)k * Fin + f) * DOUT + o];
        sp += pc[k] * w;
        sq += qc[k] * w;
    }
    WpT[(long)o * Fin + f] = (bf16_t)sp;
    Wq[(long)f * DOUT + o] = sq;
}

// s[b][f] = sum_n x[b][n][f]  (layer-1 input, fp32, Fin = 256)
__global__ __launch_bounds__(256)
void nodesum_f32(const float* __restrict__ x, float* __restrict__ s)
{
    int b = blockIdx.x;
    int f = threadIdx.x;   // 256 = Fin
    float sum = 0.f;
    for (int n = 0; n < N_NODES; ++n)
        sum += x[((long)b * N_NODES + n) * 256 + f];
    s[(long)b * 256 + f] = sum;
}

// Z[b][o] = sum_f s[b][f] * Wq[f][o]
__global__ __launch_bounds__(128)
void zmat(const float* __restrict__ s, const float* __restrict__ Wq,
          float* __restrict__ Z, int K)
{
    __shared__ float sh[256];
    int b = blockIdx.x;
    int o = threadIdx.x;
    for (int i = threadIdx.x; i < K; i += 128) sh[i] = s[(long)b * K + i];
    __syncthreads();
    float acc = 0.f;
    for (int f = 0; f < K; ++f) acc = fmaf(sh[f], Wq[(long)f * DOUT + o], acc);
    Z[(long)b * DOUT + o] = acc;
}

// ---------------------------------------------------------------------------
// Main WMMA GEMM:  H[m][o] = sum_k A[m][k] * WpT[o][k] + Z[m/64][o]
// A: bf16 (or fp32 x for layer 1), row-major MxK.  WpT: K-major 128xK.
// Block = 256 threads = 8 waves; each wave owns TWO 16-row M tiles x all 128 N
// (each B fragment feeds 2 WMMAs; B is software-pipelined through registers).
// ---------------------------------------------------------------------------
template<int K, bool AF32>
__global__ __launch_bounds__(256)
void gemm_wmma(const void* __restrict__ A_, const bf16_t* __restrict__ Bt,
               const float* __restrict__ Z, bf16_t* __restrict__ Hout)
{
    constexpr int LDSK = K + 8;                 // pad: conflict-free ds_read_b128
    __shared__ bf16_t ldsB[DOUT * LDSK];

    // stage transposed weight into LDS (16B chunks)
    {
        constexpr int CPR = K / 8;              // uint4 chunks per row
        for (int idx = threadIdx.x; idx < DOUT * CPR; idx += 256) {
            int r = idx / CPR, c = idx % CPR;
            ((uint4*)(ldsB + (long)r * LDSK))[c] =
                ((const uint4*)(Bt + (long)r * K))[c];
        }
    }
    __syncthreads();

    const int lane  = threadIdx.x & 31;
    const int wave  = threadIdx.x >> 5;
    const int row16 = lane & 15;
    const int half  = lane >> 4;
    const long m0   = (long)blockIdx.x * 256 + wave * 32;  // wave's 32-row range
    const long ar0  = m0 + row16;        // A row, m-tile 0
    const long ar1  = m0 + 16 + row16;   // A row, m-tile 1

    v8f acc0[8], acc1[8];
#pragma unroll
    for (int nt = 0; nt < 8; ++nt) { acc0[nt] = (v8f)(0.f); acc1[nt] = (v8f)(0.f); }

    const bf16_t* __restrict__ Ab = (const bf16_t*)A_;
    const float*  __restrict__ Af = (const float*)A_;

#pragma unroll
    for (int kt = 0; kt < K; kt += 32) {
        // ---- A fragments: 16 bf16 = [K: kt+half*8 .. +8] ++ [kt+16+half*8 .. +8]
        union U { v16bf v; v8bf h[2]; __bf16 e[16]; };
        U ua0, ua1;
        if constexpr (AF32) {
            const float* p0 = Af + ar0 * K + kt + half * 8;
            const float* p1 = Af + ar1 * K + kt + half * 8;
            v4f a0 = *(const v4f*)(p0),      a1 = *(const v4f*)(p0 + 4);
            v4f a2 = *(const v4f*)(p0 + 16), a3 = *(const v4f*)(p0 + 20);
            v4f c0 = *(const v4f*)(p1),      c1 = *(const v4f*)(p1 + 4);
            v4f c2 = *(const v4f*)(p1 + 16), c3 = *(const v4f*)(p1 + 20);
#pragma unroll
            for (int i = 0; i < 4; ++i) {
                ua0.e[i]      = (__bf16)a0[i];  ua0.e[4 + i]  = (__bf16)a1[i];
                ua0.e[8 + i]  = (__bf16)a2[i];  ua0.e[12 + i] = (__bf16)a3[i];
                ua1.e[i]      = (__bf16)c0[i];  ua1.e[4 + i]  = (__bf16)c1[i];
                ua1.e[8 + i]  = (__bf16)c2[i];  ua1.e[12 + i] = (__bf16)c3[i];
            }
        } else {
            const bf16_t* p0 = Ab + ar0 * K + kt + half * 8;
            const bf16_t* p1 = Ab + ar1 * K + kt + half * 8;
            ua0.h[0] = *(const v8bf*)(p0);  ua0.h[1] = *(const v8bf*)(p0 + 16);
            ua1.h[0] = *(const v8bf*)(p1);  ua1.h[1] = *(const v8bf*)(p1 + 16);
        }

        // ---- B fragments from LDS, software-pipelined: load nt+1 during nt's WMMAs
        const bf16_t* brow = ldsB + (long)row16 * LDSK + kt + half * 8;
        U ub, ubn;
        ub.h[0] = *(const v8bf*)(brow);
        ub.h[1] = *(const v8bf*)(brow + 16);
#pragma unroll
        for (int nt = 0; nt < 8; ++nt) {
            if (nt < 7) {
                const bf16_t* bp = brow + (long)(nt + 1) * 16 * LDSK;
                ubn.h[0] = *(const v8bf*)(bp);
                ubn.h[1] = *(const v8bf*)(bp + 16);
            }
            acc0[nt] = __builtin_amdgcn_wmma_f32_16x16x32_bf16(
                false, ua0.v, false, ub.v, (short)0, acc0[nt], false, false);
            acc1[nt] = __builtin_amdgcn_wmma_f32_16x16x32_bf16(
                false, ua1.v, false, ub.v, (short)0, acc1[nt], false, false);
            ub = ubn;
        }
    }

    // ---- epilogue: + Z[batch][o], store bf16 (16-row tile never crosses batch)
#pragma unroll
    for (int mt = 0; mt < 2; ++mt) {
        const long mb   = m0 + mt * 16;
        const long bidx = mb >> 6;
        const v8f* acc  = mt ? acc1 : acc0;
#pragma unroll
        for (int nt = 0; nt < 8; ++nt) {
            int o = nt * 16 + row16;
            float z = Z[bidx * DOUT + o];
#pragma unroll
            for (int r = 0; r < 8; ++r) {
                long m = mb + half * 8 + r;
                Hout[m * DOUT + o] = (bf16_t)(acc[nt][r] + z);
            }
        }
    }
}

// per-feature sum / sumsq accumulation (BN stats), acc[256] pre-zeroed
__global__ __launch_bounds__(256)
void stats_kernel(const bf16_t* __restrict__ H, float* __restrict__ acc, int M)
{
    int o  = threadIdx.x & 127;
    int h2 = threadIdx.x >> 7;
    float s = 0.f, ss = 0.f;
    for (long m = (long)blockIdx.x * 2 + h2; m < M; m += (long)gridDim.x * 2) {
        float v = (float)H[m * DOUT + o];
        s += v; ss += v * v;
    }
    __shared__ float sh[256];
    sh[threadIdx.x] = s;  __syncthreads();
    if (h2 == 0) atomicAdd(&acc[o], s + sh[128 + o]);
    __syncthreads();
    sh[threadIdx.x] = ss; __syncthreads();
    if (h2 == 0) atomicAdd(&acc[128 + o], ss + sh[128 + o]);
}

__global__ void zero_stats(float* acc) { acc[threadIdx.x] = 0.f; }

__global__ void finalize_stats(const float* __restrict__ acc,
                               float* __restrict__ murstd, float Minv)
{
    int o = threadIdx.x;  // 128
    float mu  = acc[o] * Minv;
    float var = acc[128 + o] * Minv - mu * mu;
    murstd[o]       = mu;
    murstd[128 + o] = rsqrtf(var + 1e-5f);
}

// A'[m][o] = relu(g*(H-mu)*rstd + b) (bf16) and s[b][o] = sum_n A'
__global__ __launch_bounds__(256)
void act_nodesum(const bf16_t* __restrict__ H, const float* __restrict__ murstd,
                 const float* __restrict__ g, const float* __restrict__ bta,
                 bf16_t* __restrict__ Aout, float* __restrict__ s)
{
    int b  = blockIdx.x;
    int o  = threadIdx.x & 127;
    int h2 = threadIdx.x >> 7;
    float mu = murstd[o], rstd = murstd[128 + o];
    float gg = g[o], bb = bta[o];
    float sum = 0.f;
    for (int i = 0; i < 32; ++i) {
        int n = h2 + 2 * i;
        long idx = ((long)b * N_NODES + n) * DOUT + o;
        float a = fmaxf(0.f, gg * ((float)H[idx] - mu) * rstd + bb);
        Aout[idx] = (bf16_t)a;
        sum += a;
    }
    __shared__ float sh[256];
    sh[threadIdx.x] = sum; __syncthreads();
    if (h2 == 0) s[(long)b * DOUT + o] = sum + sh[128 + o];
}

// out[b][o] = mean_n relu(g*(H-mu)*rstd + b)
__global__ __launch_bounds__(256)
void final_pool(const bf16_t* __restrict__ H, const float* __restrict__ murstd,
                const float* __restrict__ g, const float* __restrict__ bta,
                float* __restrict__ out)
{
    int b  = blockIdx.x;
    int o  = threadIdx.x & 127;
    int h2 = threadIdx.x >> 7;
    float mu = murstd[o], rstd = murstd[128 + o];
    float gg = g[o], bb = bta[o];
    float sum = 0.f;
    for (int i = 0; i < 32; ++i) {
        int n = h2 + 2 * i;
        long idx = ((long)b * N_NODES + n) * DOUT + o;
        sum += fmaxf(0.f, gg * ((float)H[idx] - mu) * rstd + bb);
    }
    __shared__ float sh[256];
    sh[threadIdx.x] = sum; __syncthreads();
    if (h2 == 0) out[(long)b * DOUT + o] = (sum + sh[128 + o]) * (1.f / N_NODES);
}

extern "C" void kernel_launch(void* const* d_in, const int* in_sizes, int n_in,
                              void* d_out, int out_size, void* d_ws, size_t ws_size,
                              hipStream_t stream)
{
    const float* x  = (const float*)d_in[0];
    const float* W1 = (const float*)d_in[1];
    const float* W2 = (const float*)d_in[2];
    const float* W3 = (const float*)d_in[3];
    const float* g1 = (const float*)d_in[4];
    const float* b1 = (const float*)d_in[5];
    const float* g2 = (const float*)d_in[6];
    const float* b2 = (const float*)d_in[7];
    const float* g3 = (const float*)d_in[8];
    const float* b3 = (const float*)d_in[9];
    float* out = (float*)d_out;

    // Lhat = alpha*I + beta*J  (exact closed form of the reference's graph)
    double stdv  = sqrt(63.0 / 4095.0);
    double ao    = exp(-1.0 / stdv);
    double deg   = 1.0 + 63.0 * ao;
    double alpha = -(1.0 - ao) / deg;
    double beta  = -ao / deg;
    double gamma = alpha + 64.0 * beta;
    double p[5] = {1.0, alpha, 0, 0, 0}, q[5] = {0.0, beta, 0, 0, 0};
    for (int k = 2; k < 5; ++k) {
        p[k] = 2.0 * alpha * p[k - 1] - p[k - 2];
        q[k] = 2.0 * (beta * p[k - 1] + gamma * q[k - 1]) - q[k - 2];
    }
    float pf[5], qf[5];
    for (int k = 0; k < 5; ++k) { pf[k] = (float)p[k]; qf[k] = (float)q[k]; }

    // workspace layout
    char* ws = (char*)d_ws;
    bf16_t* WpT1 = (bf16_t*)(ws);                      //  64 KB (128 x 256)
    bf16_t* WpT2 = (bf16_t*)(ws + (64 << 10));         //  32 KB
    bf16_t* WpT3 = (bf16_t*)(ws + (96 << 10));         //  32 KB
    float*  Wq1  = (float*)(ws + (128 << 10));         // 128 KB
    float*  Wq2  = (float*)(ws + (256 << 10));         //  64 KB
    float*  Wq3  = (float*)(ws + (320 << 10));         //  64 KB
    float*  sAcc = (float*)(ws + (384 << 10));         //   1 KB (sum|sumsq)
    float*  muRs = (float*)(ws + (388 << 10));         //   1 KB (mu|rstd)
    float*  S    = (float*)(ws + ((size_t)1 << 20));   //   4 MB node sums
    float*  Z    = (float*)(ws + ((size_t)6 << 20));   //   2 MB
    bf16_t* H    = (bf16_t*)(ws + ((size_t)8 << 20));  //  64 MB layer output
    bf16_t* A    = (bf16_t*)(ws + ((size_t)72 << 20)); //  64 MB activated input

    const int M = 4096 * N_NODES;  // 262144
    const float Minv = 1.0f / (float)M;

    // weight prep
    prep_weights<<<(256 * 128 + 255) / 256, 256, 0, stream>>>(
        W1, WpT1, Wq1, 5, 256, pf[0], pf[1], pf[2], pf[3], pf[4],
        qf[0], qf[1], qf[2], qf[3], qf[4]);
    prep_weights<<<(128 * 128 + 255) / 256, 256, 0, stream>>>(
        W2, WpT2, Wq2, 3, 128, pf[0], pf[1], pf[2], pf[3], pf[4],
        qf[0], qf[1], qf[2], qf[3], qf[4]);
    prep_weights<<<(128 * 128 + 255) / 256, 256, 0, stream>>>(
        W3, WpT3, Wq3, 3, 128, pf[0], pf[1], pf[2], pf[3], pf[4],
        qf[0], qf[1], qf[2], qf[3], qf[4]);

    // ---- layer 1 (x fp32, K=256)
    nodesum_f32<<<4096, 256, 0, stream>>>(x, S);
    zmat<<<4096, 128, 0, stream>>>(S, Wq1, Z, 256);
    gemm_wmma<256, true><<<M / 256, 256, 0, stream>>>(x, WpT1, Z, H);
    zero_stats<<<1, 256, 0, stream>>>(sAcc);
    stats_kernel<<<1024, 256, 0, stream>>>(H, sAcc, M);
    finalize_stats<<<1, 128, 0, stream>>>(sAcc, muRs, Minv);
    act_nodesum<<<4096, 256, 0, stream>>>(H, muRs, g1, b1, A, S);

    // ---- layer 2 (K=128)
    zmat<<<4096, 128, 0, stream>>>(S, Wq2, Z, 128);
    gemm_wmma<128, false><<<M / 256, 256, 0, stream>>>(A, WpT2, Z, H);
    zero_stats<<<1, 256, 0, stream>>>(sAcc);
    stats_kernel<<<1024, 256, 0, stream>>>(H, sAcc, M);
    finalize_stats<<<1, 128, 0, stream>>>(sAcc, muRs, Minv);
    act_nodesum<<<4096, 256, 0, stream>>>(H, muRs, g2, b2, A, S);

    // ---- layer 3 (K=128)
    zmat<<<4096, 128, 0, stream>>>(S, Wq3, Z, 128);
    gemm_wmma<128, false><<<M / 256, 256, 0, stream>>>(A, WpT3, Z, H);
    zero_stats<<<1, 256, 0, stream>>>(sAcc);
    stats_kernel<<<1024, 256, 0, stream>>>(H, sAcc, M);
    finalize_stats<<<1, 128, 0, stream>>>(sAcc, muRs, Minv);
    final_pool<<<4096, 256, 0, stream>>>(H, muRs, g3, b3, out);

    (void)in_sizes; (void)n_in; (void)out_size; (void)ws_size;
}
